// MultiHeadSelfAttention_43774306681025
// MI455X (gfx1250) — matrix operand order repaired and checked by reading, not compiled, not run on hardware
//
#include <hip/hip_runtime.h>

// MI455X / gfx1250 wave32 WMMA implementation of multi-head self attention.
// fp32 in/out, f16 WMMA (16x16x32) with fp32 accumulation, flash-attention
// style streaming softmax (no materialized score matrix).

typedef _Float16 half_t;
typedef __attribute__((ext_vector_type(16))) _Float16 v16h;
typedef __attribute__((ext_vector_type(8)))  _Float16 v8h;
typedef __attribute__((ext_vector_type(8)))  float    v8f;

union V16HU { v16h v; v8h h[2]; };

#define L_SEQ 2048
#define DM    1024
#define NH    16
#define DK    64

// ---------------------------------------------------------------------------
// Fragment loaders (CDNA5 ISA 7.12.2, wave32)
// A-matrix 16x32 f16: lane r=lane&15 holds row M=r; VGPR0-3 = K kb..kb+7,
// VGPR4-7 = K kb+16..kb+23, kb = 8*(lane>=16).  Source: row-major, leading dim ld.
__device__ __forceinline__ v16h load_frag_a(const half_t* __restrict__ p0, int ld) {
  const int lane = threadIdx.x & 31;
  const int r  = lane & 15;
  const int kb = (lane >> 4) << 3;
  const half_t* p = p0 + r * ld + kb;
  V16HU u;
  u.h[0] = *(const v8h*)(p);
  u.h[1] = *(const v8h*)(p + 16);
  return u.v;
}

// B-matrix 32x16 f16: lane n=lane&15 holds column N=n; VGPR0-3 = K koff..koff+7,
// VGPR4-7 = K koff+16..koff+23, koff = 8*(lane>=16).  Source stored N x K row-major (i.e. B^T / W rows).
__device__ __forceinline__ v16h load_frag_b(const half_t* __restrict__ p0, int ld) {
  const int lane = threadIdx.x & 31;
  const int n    = lane & 15;
  const int koff = (lane >> 4) << 3;
  const half_t* p = p0 + n * ld + koff;
  V16HU u;
  u.h[0] = *(const v8h*)(p);
  u.h[1] = *(const v8h*)(p + 16);
  return u.v;
}

__device__ __forceinline__ v8f wmma16(v16h a, v16h b, v8f c) {
  return __builtin_amdgcn_wmma_f32_16x16x32_f16(false, a, false, b, (short)0, c,
                                                false, false);
}

// ---------------------------------------------------------------------------
// fp32 -> f16 conversion
__global__ void mha_cvt_f16(const float* __restrict__ in, half_t* __restrict__ out,
                            int n) {
  int i = blockIdx.x * blockDim.x + threadIdx.x;
  int stride = gridDim.x * blockDim.x;
  for (; i < n; i += stride) out[i] = (half_t)in[i];
}

// ---------------------------------------------------------------------------
// GEMM: C[M,1024] = A[M,1024](f16) x W^T + bias, W is [N=1024, K=1024] row-major f16.
// Block = 128 threads (4 waves); each wave computes a 32(M) x 64(N) tile with
// software-pipelined (double-buffered) fragment loads so WMMAs wait only on the
// previous iteration's load group instead of loadcnt==0.
// mode 0: Q -> f16 [B,H,L,DK], scaled by 1/sqrt(DK)
// mode 1: K -> f16 [B,H,L,DK]
// mode 2: V -> f16 transposed [B,H,DK,L]
// mode 3: final out -> fp32 [M,1024]
__global__ __launch_bounds__(128) void mha_gemm(
    const half_t* __restrict__ A, const half_t* __restrict__ W,
    const float* __restrict__ bias, void* __restrict__ out, int mode) {
  const int wave = threadIdx.x >> 5;
  const int lane = threadIdx.x & 31;
  const int mbase = blockIdx.x * 32;
  const int nbase = blockIdx.y * 256 + wave * 64;

  const half_t* A0 = A + (size_t)mbase * DM;
  const half_t* A1 = A + (size_t)(mbase + 16) * DM;
  const half_t* W0 = W + (size_t)nbase * DM;

  v8f acc[2][4] = {};

  // prologue: fragments for k = 0
  v16h a0 = load_frag_a(A0, DM);
  v16h a1 = load_frag_a(A1, DM);
  v16h b[4];
#pragma unroll
  for (int t = 0; t < 4; ++t) b[t] = load_frag_b(W0 + (size_t)(16 * t) * DM, DM);

#pragma unroll 2
  for (int k = 32; k < DM; k += 32) {
    // issue next iteration's loads first (stay in flight behind the WMMAs)
    v16h na0 = load_frag_a(A0 + k, DM);
    v16h na1 = load_frag_a(A1 + k, DM);
    v16h nb[4];
#pragma unroll
    for (int t = 0; t < 4; ++t)
      nb[t] = load_frag_b(W0 + (size_t)(16 * t) * DM + k, DM);

#pragma unroll
    for (int t = 0; t < 4; ++t) {
      acc[0][t] = wmma16(a0, b[t], acc[0][t]);
      acc[1][t] = wmma16(a1, b[t], acc[1][t]);
    }
    a0 = na0; a1 = na1;
#pragma unroll
    for (int t = 0; t < 4; ++t) b[t] = nb[t];
  }
  // epilogue k-step
#pragma unroll
  for (int t = 0; t < 4; ++t) {
    acc[0][t] = wmma16(a0, b[t], acc[0][t]);
    acc[1][t] = wmma16(a1, b[t], acc[1][t]);
  }

  // C/D layout: lane n = lane&15 is column; VGPR v is row v (+8 for upper half).
  const int rowoff = (lane >> 4) << 3;
  const int ncol = lane & 15;
#pragma unroll
  for (int s = 0; s < 2; ++s) {
#pragma unroll
    for (int t = 0; t < 4; ++t) {
      const int n = nbase + 16 * t + ncol;
      const float bv = bias[n];
#pragma unroll
      for (int v = 0; v < 8; ++v) {
        const int m = mbase + 16 * s + rowoff + v;
        float val = acc[s][t][v] + bv;
        if (mode == 3) {
          ((float*)out)[(size_t)m * DM + n] = val;
        } else {
          const int bb = m >> 11, l = m & (L_SEQ - 1);
          const int h = n >> 6, dk = n & (DK - 1);
          if (mode == 0) val *= 0.125f;  // 1/sqrt(64)
          size_t idx;
          if (mode == 2)  // V transposed: [B,H,DK,L]
            idx = ((size_t)(bb * NH + h) * DK + dk) * L_SEQ + l;
          else            // Q/K: [B,H,L,DK]
            idx = ((size_t)(bb * NH + h) * L_SEQ + l) * DK + dk;
          ((half_t*)out)[idx] = (half_t)val;
        }
      }
    }
  }
}

// ---------------------------------------------------------------------------
// Flash attention. Block = 4 waves; wave w handles q-tile [qbase, qbase+16) of
// head (b,h).  Streams keys in blocks of 32, online softmax, O accumulated in
// fp32 WMMA fragments, P transposed C-layout -> A-layout through LDS.
__global__ __launch_bounds__(128) void mha_flash(
    const half_t* __restrict__ Qh,   // [B,H,L,DK] (pre-scaled by 1/sqrt(dk))
    const half_t* __restrict__ Kh,   // [B,H,L,DK]
    const half_t* __restrict__ Vt,   // [B,H,DK,L]
    const int* __restrict__ mask,    // [B,L]
    half_t* __restrict__ Oh) {       // [B,L,H*DK]
  __shared__ half_t pbuf[4][16 * 32];
  const int wave = threadIdx.x >> 5;
  const int lane = threadIdx.x & 31;
  const int bh = blockIdx.y;
  const int b = bh >> 4, h = bh & 15;
  const int qbase = blockIdx.x * 64 + wave * 16;

  const half_t* Qhead = Qh + (size_t)bh * L_SEQ * DK;
  const half_t* Khead = Kh + (size_t)bh * L_SEQ * DK;
  const half_t* Vhead = Vt + (size_t)bh * DK * L_SEQ;
  const int* mrow = mask + b * L_SEQ;
  half_t* plds = pbuf[wave];

  // Q fragments for this tile: d 0..31 and 32..63
  const v16h qa0 = load_frag_a(Qhead + (size_t)qbase * DK + 0, DK);
  const v16h qa1 = load_frag_a(Qhead + (size_t)qbase * DK + 32, DK);

  v8f o[4] = {v8f{}, v8f{}, v8f{}, v8f{}};
  float run_m[8], run_l[8];
#pragma unroll
  for (int v = 0; v < 8; ++v) { run_m[v] = -3.0e38f; run_l[v] = 0.0f; }

  const int ncol = lane & 15;
  const int rowoff = (lane >> 4) << 3;

  for (int kb = 0; kb < L_SEQ; kb += 32) {
    if (kb + 32 < L_SEQ) {  // gfx1250 global_prefetch of next K/V key block
      __builtin_prefetch(Khead + (size_t)(kb + 32) * DK + lane * 2, 0, 0);
      __builtin_prefetch(Vhead + (size_t)lane * L_SEQ + kb + 32, 0, 0);
    }

    // ---- S = Q K^T for 2 key tiles (16 keys each), K-dim = 64 in two steps
    v8f s0 = {}, s1 = {};
    s0 = wmma16(qa0, load_frag_b(Khead + (size_t)(kb +  0) * DK + 0,  DK), s0);
    s0 = wmma16(qa1, load_frag_b(Khead + (size_t)(kb +  0) * DK + 32, DK), s0);
    s1 = wmma16(qa0, load_frag_b(Khead + (size_t)(kb + 16) * DK + 0,  DK), s1);
    s1 = wmma16(qa1, load_frag_b(Khead + (size_t)(kb + 16) * DK + 32, DK), s1);

    // ---- mask (scores already scaled via Q): lane's key per tile
    if (mrow[kb + ncol] == 0) {
#pragma unroll
      for (int v = 0; v < 8; ++v) s0[v] = -1.0e9f;
    }
    if (mrow[kb + 16 + ncol] == 0) {
#pragma unroll
      for (int v = 0; v < 8; ++v) s1[v] = -1.0e9f;
    }

    // ---- online softmax: row = (v, lane-half); reduce over 16 lanes per half
    float tmax[8];
#pragma unroll
    for (int v = 0; v < 8; ++v) tmax[v] = fmaxf(s0[v], s1[v]);
#pragma unroll
    for (int off = 8; off >= 1; off >>= 1)
#pragma unroll
      for (int v = 0; v < 8; ++v)
        tmax[v] = fmaxf(tmax[v], __shfl_xor(tmax[v], off, 32));

    float p0[8], p1[8], rsum[8], alpha[8];
#pragma unroll
    for (int v = 0; v < 8; ++v) {
      const float nm = fmaxf(run_m[v], tmax[v]);
      alpha[v] = __expf(run_m[v] - nm);
      run_m[v] = nm;
      p0[v] = __expf(s0[v] - nm);
      p1[v] = __expf(s1[v] - nm);
      rsum[v] = p0[v] + p1[v];
    }
#pragma unroll
    for (int off = 8; off >= 1; off >>= 1)
#pragma unroll
      for (int v = 0; v < 8; ++v) rsum[v] += __shfl_xor(rsum[v], off, 32);
#pragma unroll
    for (int v = 0; v < 8; ++v) run_l[v] = run_l[v] * alpha[v] + rsum[v];
#pragma unroll
    for (int t = 0; t < 4; ++t)
#pragma unroll
      for (int v = 0; v < 8; ++v) o[t][v] *= alpha[v];

    // ---- transpose P (C-layout) -> A-layout via LDS (wave-private 1KB)
#pragma unroll
    for (int v = 0; v < 8; ++v) {
      plds[(v + rowoff) * 32 + ncol]      = (half_t)p0[v];
      plds[(v + rowoff) * 32 + ncol + 16] = (half_t)p1[v];
    }
    asm volatile("s_wait_dscnt 0" ::: "memory");  // wave-local LDS RAW fence
    const v16h pf = load_frag_a(plds, 32);

    // ---- O += P x V  (V stored [d, key]; 4 d-tiles of 16)
#pragma unroll
    for (int t = 0; t < 4; ++t)
      o[t] = wmma16(pf, load_frag_b(Vhead + (size_t)(t * 16) * L_SEQ + kb, L_SEQ),
                    o[t]);
  }

  // ---- finalize: divide by running sum, write f16 [B,L,H*DK]
#pragma unroll
  for (int v = 0; v < 8; ++v) {
    const float inv = 1.0f / run_l[v];
    const int q = qbase + rowoff + v;
    const size_t base = ((size_t)b * L_SEQ + q) * DM + h * DK;
#pragma unroll
    for (int t = 0; t < 4; ++t)
      Oh[base + t * 16 + ncol] = (half_t)(o[t][v] * inv);
  }
}

// ---------------------------------------------------------------------------
extern "C" void kernel_launch(void* const* d_in, const int* in_sizes, int n_in,
                              void* d_out, int out_size, void* d_ws, size_t ws_size,
                              hipStream_t stream) {
  const float* x    = (const float*)d_in[0];
  const int*   mask = (const int*)d_in[1];
  const float* Wq = (const float*)d_in[2]; const float* bq = (const float*)d_in[3];
  const float* Wk = (const float*)d_in[4]; const float* bk = (const float*)d_in[5];
  const float* Wv = (const float*)d_in[6]; const float* bv = (const float*)d_in[7];
  const float* Wo = (const float*)d_in[8]; const float* bo = (const float*)d_in[9];

  const int M = 4 * L_SEQ;              // 8192 tokens
  char* ws = (char*)d_ws;
  const size_t MB = 1ull << 20;
  half_t* xh  = (half_t*)(ws + 0 * MB);   // 16 MB  x in f16
  half_t* wqh = (half_t*)(ws + 16 * MB);  //  2 MB
  half_t* wkh = (half_t*)(ws + 18 * MB);
  half_t* wvh = (half_t*)(ws + 20 * MB);
  half_t* woh = (half_t*)(ws + 22 * MB);
  half_t* qh  = (half_t*)(ws + 24 * MB);  // 16 MB  [B,H,L,DK]
  half_t* kh  = (half_t*)(ws + 40 * MB);  // 16 MB  [B,H,L,DK]
  half_t* vt  = (half_t*)(ws + 56 * MB);  // 16 MB  [B,H,DK,L]
  half_t* ah  = (half_t*)(ws + 72 * MB);  // 16 MB  attention out [B,L,H*DK]
                                          // total 88 MB

  // fp32 -> f16 conversions
  mha_cvt_f16<<<4096, 256, 0, stream>>>(x, xh, M * DM);
  mha_cvt_f16<<<1024, 256, 0, stream>>>(Wq, wqh, DM * DM);
  mha_cvt_f16<<<1024, 256, 0, stream>>>(Wk, wkh, DM * DM);
  mha_cvt_f16<<<1024, 256, 0, stream>>>(Wv, wvh, DM * DM);
  mha_cvt_f16<<<1024, 256, 0, stream>>>(Wo, woh, DM * DM);

  // Q/K/V projections (WMMA), writing head-split (and V-transposed) f16
  dim3 gg(M / 32, DM / 256);
  mha_gemm<<<gg, 128, 0, stream>>>(xh, wqh, bq, qh, 0);
  mha_gemm<<<gg, 128, 0, stream>>>(xh, wkh, bk, kh, 1);
  mha_gemm<<<gg, 128, 0, stream>>>(xh, wvh, bv, vt, 2);

  // flash attention: grid = (L/64 q-blocks, B*H heads), 4 waves/block
  mha_flash<<<dim3(L_SEQ / 64, 4 * NH), 128, 0, stream>>>(qh, kh, vt, mask, ah);

  // output projection -> fp32 d_out
  mha_gemm<<<gg, 128, 0, stream>>>(ah, woh, bo, d_out, 3);
}
